// PointNetSetAbstraction_55688545960047
// MI455X (gfx1250) — compile-verified, matrix operand order checked by reading, not run on hardware
//
#include <hip/hip_runtime.h>
#include <hip/hip_bf16.h>

typedef __attribute__((ext_vector_type(16))) _Float16 v16h;
typedef __attribute__((ext_vector_type(8)))  _Float16 v8h;
typedef __attribute__((ext_vector_type(8)))  float    v8f;

#define NPTS   4096
#define NCENT  1024
#define KSAMP  32
#define BATCH  8
#define MTOT   (BATCH * NCENT * KSAMP)   // 262144
#define RAD2   0.16f
#define BN_EPS 1e-5f

// Tiled activation layout: element (m, c) of a C-channel tensor lives at
//   T = m>>4, ln = m&15, c8 = c>>3, e = c&7
//   idx = ((T*(C/8) + c8)*16 + ln)*8 + e
// This makes every WMMA B-fragment half one aligned 16-byte v8h load and
// every epilogue store one aligned 16-byte v8h store, coalesced across lanes.

// ---------------------------------------------------------------------------
// Kernel 1: farthest point sampling (one workgroup per batch, xyz in LDS,
// per-thread running-min distances in registers, shuffle+LDS argmax tree).
// Also emits new_xyz both to workspace and to d_out part 1 ((B,3,S) layout).
// ---------------------------------------------------------------------------
__global__ void fps_kernel(const float* __restrict__ xyz,
                           float* __restrict__ newXyz,   // [B*S][3]
                           float* __restrict__ outXyz) { // d_out, (B,3,S)
  __shared__ float lx[NPTS], ly[NPTS], lz[NPTS];
  __shared__ float rmx[8];
  __shared__ int   rag[8];
  __shared__ int   sFar;

  const int b   = blockIdx.x;
  const int tid = threadIdx.x;
  const float* base = xyz + (long)b * 3 * NPTS;
  for (int n = tid; n < NPTS; n += 256) {
    lx[n] = base[n];
    ly[n] = base[NPTS + n];
    lz[n] = base[2 * NPTS + n];
  }
  float dist[16];
#pragma unroll
  for (int j = 0; j < 16; ++j) dist[j] = 1e10f;
  if (tid == 0) sFar = 0;
  __syncthreads();

  const int lane = tid & 31;
  const int wid  = tid >> 5;

  for (int i = 0; i < NCENT; ++i) {
    const int far = sFar;
    const float cx = lx[far], cy = ly[far], cz = lz[far];
    if (tid == 0) {
      long bs = (long)b * NCENT + i;
      newXyz[bs * 3 + 0] = cx;
      newXyz[bs * 3 + 1] = cy;
      newXyz[bs * 3 + 2] = cz;
      outXyz[(long)b * 3 * NCENT + 0 * NCENT + i] = cx;
      outXyz[(long)b * 3 * NCENT + 1 * NCENT + i] = cy;
      outXyz[(long)b * 3 * NCENT + 2 * NCENT + i] = cz;
    }
    float bd = -1.0f;
    int   bi = 0;
#pragma unroll
    for (int j = 0; j < 16; ++j) {
      const int n = tid + j * 256;
      const float dx = lx[n] - cx, dy = ly[n] - cy, dz = lz[n] - cz;
      const float d  = dx * dx + dy * dy + dz * dz;
      const float nd = dist[j] < d ? dist[j] : d;
      dist[j] = nd;
      if (nd > bd) { bd = nd; bi = n; }
    }
    // wave32 reduction, argmax with first-index tie-break
#pragma unroll
    for (int off = 16; off >= 1; off >>= 1) {
      const float od = __shfl_xor(bd, off);
      const int   oi = __shfl_xor(bi, off);
      if (od > bd || (od == bd && oi < bi)) { bd = od; bi = oi; }
    }
    if (lane == 0) { rmx[wid] = bd; rag[wid] = bi; }
    __syncthreads();
    if (tid == 0) {
      float fb = rmx[0]; int fn = rag[0];
      for (int w = 1; w < 8; ++w)
        if (rmx[w] > fb || (rmx[w] == fb && rag[w] < fn)) { fb = rmx[w]; fn = rag[w]; }
      sFar = fn;
    }
    __syncthreads();
  }
}

// ---------------------------------------------------------------------------
// Kernel 2: ball query + gather + concat. One wave per centroid, ballot-based
// compaction (first 32 in-radius indices in ascending order). Emits f16
// features directly in the WMMA-tiled layout (96 channels: 3 rel-xyz + 64
// feats + 29 zero pad).
// ---------------------------------------------------------------------------
__global__ void ballquery_group_kernel(const float* __restrict__ xyz,
                                       const float* __restrict__ pts,
                                       const float* __restrict__ newXyz,
                                       _Float16* __restrict__ X0) {
  __shared__ float lx[NPTS], ly[NPTS], lz[NPTS];
  __shared__ int sidx[8][KSAMP];
  const int b     = blockIdx.y;
  const int chunk = blockIdx.x;   // 0..15, 64 centroids each
  const int tid   = threadIdx.x;
  const float* base = xyz + (long)b * 3 * NPTS;
  for (int n = tid; n < NPTS; n += 256) {
    lx[n] = base[n];
    ly[n] = base[NPTS + n];
    lz[n] = base[2 * NPTS + n];
  }
  __syncthreads();

  const int lane = tid & 31;
  const int w    = tid >> 5;

  for (int j = 0; j < 8; ++j) {
    const int s = chunk * 64 + w * 8 + j;
    const long bs = (long)b * NCENT + s;
    const float cx = newXyz[bs * 3 + 0];
    const float cy = newXyz[bs * 3 + 1];
    const float cz = newXyz[bs * 3 + 2];

    int cnt = 0;
    for (int n0 = 0; n0 < NPTS && cnt < KSAMP; n0 += 32) {
      const int n = n0 + lane;
      const float dx = lx[n] - cx, dy = ly[n] - cy, dz = lz[n] - cz;
      const float d2 = dx * dx + dy * dy + dz * dz;
      const bool within = d2 <= RAD2;
      const unsigned mask = (unsigned)__ballot(within);
      const int slot = cnt + __popc(mask & ((1u << lane) - 1u));
      if (within && slot < KSAMP) sidx[w][slot] = n;
      cnt += __popc(mask);
    }
    __threadfence_block();
    if (cnt > KSAMP) cnt = KSAMP;
    const int first = (cnt > 0) ? sidx[w][0] : 0;
    const int n = (lane < cnt) ? sidx[w][lane] : first;

    const long m  = bs * KSAMP + lane;  // this lane's position index
    const long T  = m >> 4;
    const int  ln = (int)(m & 15);
    const float rel0 = lx[n] - cx, rel1 = ly[n] - cy, rel2 = lz[n] - cz;
    const float* pb = pts + (long)b * 64 * NPTS + n;
#pragma unroll
    for (int c8 = 0; c8 < 12; ++c8) {           // 96 channels / 8
      v8h h8;
#pragma unroll
      for (int e = 0; e < 8; ++e) {
        const int c = c8 * 8 + e;               // compile-time constant
        float val;
        if (c == 0)      val = rel0;
        else if (c == 1) val = rel1;
        else if (c == 2) val = rel2;
        else if (c < 67) val = pb[(long)(c - 3) * NPTS];
        else             val = 0.0f;
        h8[e] = (_Float16)val;
      }
      *reinterpret_cast<v8h*>(X0 + ((T * 12 + c8) * 16 + ln) * 8) = h8;
    }
  }
}

// ---------------------------------------------------------------------------
// Kernel 3: zero the batchnorm statistics accumulators (run every launch).
// ---------------------------------------------------------------------------
__global__ void zero_stats_kernel(float* __restrict__ stats) {
  const int i = blockIdx.x * blockDim.x + threadIdx.x;
  if (i < 512) stats[i] = 0.0f;
}

// ---------------------------------------------------------------------------
// Kernel 4: pre-pack f32 weights into zero-padded f16 WMMA A-fragments:
//   AF[((ct*NKB + kb)*32 + lane)*16 + 2v(+1)]  per CDNA5 ISA 7.12.2 layout.
// ---------------------------------------------------------------------------
__global__ void prep_weights_kernel(const float* __restrict__ W, int CinW,
                                    int NKB, int NCO,
                                    _Float16* __restrict__ AF) {
  const int t = blockIdx.x * blockDim.x + threadIdx.x;
  if (t >= NCO * NKB * 32) return;
  const int lane = t & 31;
  const int kb   = (t >> 5) % NKB;
  const int ct   = (t >> 5) / NKB;
  const int h    = lane >> 4;
  const int ln   = lane & 15;
  const int co   = ct * 16 + ln;
  _Float16* dst = AF + (long)t * 16;
#pragma unroll
  for (int v = 0; v < 8; ++v) {
    const int ci = kb * 32 + ((v < 4) ? 0 : 16) + h * 8 + (v & 3) * 2;
    dst[2 * v]     = (ci     < CinW) ? (_Float16)W[(long)co * CinW + ci]     : (_Float16)0.0f;
    dst[2 * v + 1] = (ci + 1 < CinW) ? (_Float16)W[(long)co * CinW + ci + 1] : (_Float16)0.0f;
  }
}

// ---------------------------------------------------------------------------
// Kernel 5: WMMA GEMM over tiled activations.
//   acc[ct] += AF(ct,kb) x BN_ReLU(X(T,kb))    for all kb
// One wave owns one 16-wide m tile and all NCO output-channel tiles.
// All fragment loads are contiguous aligned v8h/v16h vector loads.
// Epilogue: bias add, per-channel sum/sum^2 (shuffle + 2 atomics / 16 vals),
// tiled v8h store for the next layer.
// ---------------------------------------------------------------------------
template <int NKB, int NCO, bool BN>
__global__ void gemm_bn_kernel(const _Float16* __restrict__ AF,
                               const float* __restrict__ bias,
                               const _Float16* __restrict__ Xin,
                               const _Float16* __restrict__ SF,  // scale frags
                               const _Float16* __restrict__ TF,  // shift frags
                               _Float16* __restrict__ Yout,
                               float* __restrict__ stats) {
  constexpr int Cout  = NCO * 16;
  constexpr int CIN8  = NKB * 4;   // input chunk count per position
  constexpr int COUT8 = NCO * 2;   // output chunk count per position
  const int lane = threadIdx.x & 31;
  const int h    = lane >> 4;
  const int ln   = lane & 15;
  const long T   = (long)(blockIdx.x * blockDim.x + threadIdx.x) >> 5;

  v8f acc[NCO];
#pragma unroll
  for (int ct = 0; ct < NCO; ++ct) acc[ct] = (v8f){0, 0, 0, 0, 0, 0, 0, 0};

#pragma unroll
  for (int kb = 0; kb < NKB; ++kb) {
    // B fragment: two aligned 16B chunk loads (chunks kb*4+h and kb*4+2+h)
    const v8h lo = *reinterpret_cast<const v8h*>(
        Xin + ((T * CIN8 + kb * 4 + h) * 16 + ln) * 8);
    const v8h hi = *reinterpret_cast<const v8h*>(
        Xin + ((T * CIN8 + kb * 4 + 2 + h) * 16 + ln) * 8);
    v16h bf;
#pragma unroll
    for (int e = 0; e < 8; ++e) { bf[e] = lo[e]; bf[8 + e] = hi[e]; }
    if (BN) {
      const v16h sf = *reinterpret_cast<const v16h*>(SF + (kb * 32 + lane) * 16);
      const v16h tf = *reinterpret_cast<const v16h*>(TF + (kb * 32 + lane) * 16);
#pragma unroll
      for (int e = 0; e < 16; ++e) {
        _Float16 y = bf[e] * sf[e] + tf[e];
        bf[e] = y > (_Float16)0.0f ? y : (_Float16)0.0f;
      }
    }
#pragma unroll
    for (int ct = 0; ct < NCO; ++ct) {
      const v16h af = *reinterpret_cast<const v16h*>(
          AF + (long)((ct * NKB + kb) * 32 + lane) * 16);
      acc[ct] = __builtin_amdgcn_wmma_f32_16x16x32_f16(
          /*neg_a=*/false, af, /*neg_b=*/false, bf,
          /*c_mod=*/(short)0, acc[ct], /*reuse_a=*/false, /*reuse_b=*/false);
    }
  }

  float* sumP = stats;
  float* sqP  = stats + Cout;
#pragma unroll
  for (int ct = 0; ct < NCO; ++ct) {
    v8h oh;
#pragma unroll
    for (int r = 0; r < 8; ++r) {
      const int co = ct * 16 + 8 * h + r;
      const float val = acc[ct][r] + bias[co];
      oh[r] = (_Float16)val;
      float s = val, ss = val * val;
#pragma unroll
      for (int off = 8; off >= 1; off >>= 1) {
        s  += __shfl_xor(s, off);
        ss += __shfl_xor(ss, off);
      }
      if (ln == 0) {
        atomicAdd(sumP + co, s);
        atomicAdd(sqP + co, ss);
      }
    }
    *reinterpret_cast<v8h*>(Yout + ((T * COUT8 + ct * 2 + h) * 16 + ln) * 8) = oh;
  }
}

// ---------------------------------------------------------------------------
// Kernel 6: fold batch statistics + gamma/beta into per-channel scale/shift,
// and (optionally) pre-pack them as f16 B-fragment-ordered vectors for the
// consumer GEMM (NKB=2, Cin=64).
// ---------------------------------------------------------------------------
__global__ void bn_finalize_kernel(const float* __restrict__ stats,
                                   const float* __restrict__ g,
                                   const float* __restrict__ bt,
                                   float* __restrict__ scale,
                                   float* __restrict__ shift,
                                   int Cout,
                                   _Float16* __restrict__ SF,   // nullable
                                   _Float16* __restrict__ TF) { // nullable
  const int c = threadIdx.x;
  if (c < Cout) {
    const float invM = 1.0f / (float)MTOT;
    const float mean = stats[c] * invM;
    const float var  = stats[Cout + c] * invM - mean * mean;
    const float sc   = g[c] * rsqrtf(var + BN_EPS);
    scale[c] = sc;
    shift[c] = bt[c] - mean * sc;
  }
  __syncthreads();
  if (SF != nullptr && threadIdx.x < 64) {   // NKB=2 x 32 lanes
    const int t    = threadIdx.x;
    const int lane = t & 31;
    const int kb   = t >> 5;
    const int h    = lane >> 4;
#pragma unroll
    for (int e = 0; e < 16; ++e) {
      const int ch = kb * 32 + ((e < 8) ? 0 : 16) + h * 8 + (e & 7);
      SF[(long)t * 16 + e] = (_Float16)scale[ch];
      TF[(long)t * 16 + e] = (_Float16)shift[ch];
    }
  }
}

// ---------------------------------------------------------------------------
// Kernel 7: final BN + ReLU + max over K, write (B,128,S) to d_out part 2.
// Reads the tiled layer-2 activations.
// ---------------------------------------------------------------------------
__global__ void bn_relu_maxk_kernel(const _Float16* __restrict__ Y2,
                                    const float* __restrict__ sc,
                                    const float* __restrict__ sh,
                                    float* __restrict__ outp) {
  const int co = threadIdx.x;          // 0..127
  const long bs = blockIdx.x;          // b*S + s
  const float s = sc[co], t = sh[co];
  const int c8 = co >> 3, e = co & 7;
  float best = 0.0f;                   // ReLU outputs are >= 0
#pragma unroll 4
  for (int k = 0; k < KSAMP; ++k) {
    const long m = bs * KSAMP + k;
    const long T = m >> 4;
    const int ln = (int)(m & 15);
    float v = (float)Y2[((T * 16 + c8) * 16 + ln) * 8 + e] * s + t;
    v = v > 0.0f ? v : 0.0f;
    best = best > v ? best : v;
  }
  const int b  = (int)(bs >> 10);
  const int ss = (int)(bs & 1023);
  outp[((long)b * 128 + co) * NCENT + ss] = best;
}

// ---------------------------------------------------------------------------
extern "C" void kernel_launch(void* const* d_in, const int* in_sizes, int n_in,
                              void* d_out, int out_size, void* d_ws, size_t ws_size,
                              hipStream_t stream) {
  const float* xyz = (const float*)d_in[0];
  const float* pts = (const float*)d_in[1];
  const float* W0  = (const float*)d_in[2];
  const float* b0  = (const float*)d_in[3];
  const float* g0  = (const float*)d_in[4];
  const float* bt0 = (const float*)d_in[5];
  const float* W1  = (const float*)d_in[6];
  const float* b1  = (const float*)d_in[7];
  const float* g1  = (const float*)d_in[8];
  const float* bt1 = (const float*)d_in[9];
  const float* W2  = (const float*)d_in[10];
  const float* b2  = (const float*)d_in[11];
  const float* g2  = (const float*)d_in[12];
  const float* bt2 = (const float*)d_in[13];
  float* out = (float*)d_out;

  char* ws = (char*)d_ws;
  size_t off = 0;
  auto carve = [&](size_t bytes) -> char* {
    char* p = ws + off;
    off = (off + bytes + 255) & ~(size_t)255;
    return p;
  };
  float*    newXyz = (float*)carve((size_t)BATCH * NCENT * 3 * sizeof(float));
  float*    stats  = (float*)carve(512 * sizeof(float)); // L0@0(128), L1@128, L2@256(256)
  float*    ssbuf  = (float*)carve(512 * sizeof(float)); // per-layer scale/shift (f32)
  _Float16* AF0 = (_Float16*)carve((size_t)4 * 3 * 32 * 16 * sizeof(_Float16));
  _Float16* AF1 = (_Float16*)carve((size_t)4 * 2 * 32 * 16 * sizeof(_Float16));
  _Float16* AF2 = (_Float16*)carve((size_t)8 * 2 * 32 * 16 * sizeof(_Float16));
  _Float16* SF0 = (_Float16*)carve((size_t)64 * 16 * sizeof(_Float16));
  _Float16* TF0 = (_Float16*)carve((size_t)64 * 16 * sizeof(_Float16));
  _Float16* SF1 = (_Float16*)carve((size_t)64 * 16 * sizeof(_Float16));
  _Float16* TF1 = (_Float16*)carve((size_t)64 * 16 * sizeof(_Float16));
  _Float16* X0 = (_Float16*)carve((size_t)MTOT * 96  * sizeof(_Float16));
  _Float16* Y0 = (_Float16*)carve((size_t)MTOT * 64  * sizeof(_Float16));
  _Float16* Y1 = (_Float16*)carve((size_t)MTOT * 64  * sizeof(_Float16));
  _Float16* Y2 = (_Float16*)carve((size_t)MTOT * 128 * sizeof(_Float16));

  // 1) FPS (also writes d_out part 1: (B,3,S) centroids)
  fps_kernel<<<BATCH, 256, 0, stream>>>(xyz, newXyz, out);

  // 2) weight fragment packing (independent of FPS; tiny)
  prep_weights_kernel<<<2, 256, 0, stream>>>(W0, 67, 3, 4, AF0);
  prep_weights_kernel<<<1, 256, 0, stream>>>(W1, 64, 2, 4, AF1);
  prep_weights_kernel<<<2, 256, 0, stream>>>(W2, 64, 2, 8, AF2);
  zero_stats_kernel<<<2, 256, 0, stream>>>(stats);

  // 3) ball query + gather + concat -> X0 (f16, tiled, 96 ch)
  ballquery_group_kernel<<<dim3(16, BATCH), 256, 0, stream>>>(xyz, pts, newXyz, X0);

  const int gemmBlocks = (MTOT / 16) / 8;  // 8 waves per 256-thread block

  // 4) layer 0: 67->64 (padded to 96 = 3x K32), no input BN
  gemm_bn_kernel<3, 4, false><<<gemmBlocks, 256, 0, stream>>>(
      AF0, b0, X0, nullptr, nullptr, Y0, stats + 0);
  bn_finalize_kernel<<<1, 128, 0, stream>>>(stats + 0, g0, bt0,
                                            ssbuf + 0, ssbuf + 64, 64, SF0, TF0);

  // 5) layer 1: 64->64, fused BN0+ReLU on load
  gemm_bn_kernel<2, 4, true><<<gemmBlocks, 256, 0, stream>>>(
      AF1, b1, Y0, SF0, TF0, Y1, stats + 128);
  bn_finalize_kernel<<<1, 128, 0, stream>>>(stats + 128, g1, bt1,
                                            ssbuf + 128, ssbuf + 192, 64, SF1, TF1);

  // 6) layer 2: 64->128, fused BN1+ReLU on load
  gemm_bn_kernel<2, 8, true><<<gemmBlocks, 256, 0, stream>>>(
      AF2, b2, Y1, SF1, TF1, Y2, stats + 256);
  bn_finalize_kernel<<<1, 128, 0, stream>>>(stats + 256, g2, bt2,
                                            ssbuf + 256, ssbuf + 384, 128,
                                            nullptr, nullptr);

  // 7) BN2 + ReLU + max over K -> d_out part 2 ((B,128,S))
  bn_relu_maxk_kernel<<<BATCH * NCENT, 128, 0, stream>>>(
      Y2, ssbuf + 256, ssbuf + 384, out + BATCH * 3 * NCENT);
}
